// NONA_NN_19628000542937
// MI455X (gfx1250) — matrix operand order repaired and checked by reading, not compiled, not run on hardware
//
#include <hip/hip_runtime.h>

typedef __attribute__((ext_vector_type(16))) _Float16 v16h;
typedef __attribute__((ext_vector_type(8)))  _Float16 v8h;
typedef __attribute__((ext_vector_type(8)))  float    v8f;

#define N_ROWS 8192
#define M_ROWS 8192
#define TOT    (N_ROWS + M_ROWS)
#define D_IN   256
#define HID1   128
#define HID2   64
#define NCLS   10
#define SPLIT  16
#define BN_EPS 1e-5f

// ---------------- Layer 0: H0 = tanh(X @ W0^T + b0), X = [x ; x_n] ----------------
__global__ void k_gemm0(const float* __restrict__ x, const float* __restrict__ xn,
                        const float* __restrict__ W0, const float* __restrict__ b0,
                        float* __restrict__ H0) {
  __shared__ float row[D_IN];
  const int r = blockIdx.x;
  const float* src = (r < N_ROWS) ? (x + (size_t)r * D_IN)
                                  : (xn + (size_t)(r - N_ROWS) * D_IN);
  for (int i = threadIdx.x; i < D_IN; i += blockDim.x) row[i] = src[i];
  __syncthreads();
  const int c = threadIdx.x;                 // 128 threads = 128 output features
  const float* w = W0 + (size_t)c * D_IN;
  float acc = b0[c];
#pragma unroll 8
  for (int d = 0; d < D_IN; ++d) acc = fmaf(row[d], w[d], acc);
  H0[(size_t)r * HID1 + c] = tanhf(acc);
}

// -------- BatchNorm stats per (batch, column) -> affine scale/shift ----------------
__global__ void k_stats(const float* __restrict__ H, int width,
                        const float* __restrict__ g, const float* __restrict__ bt,
                        float* __restrict__ scale, float* __restrict__ shift) {
  const int col   = blockIdx.x % width;
  const int batch = blockIdx.x / width;
  const float* base = H + (size_t)batch * N_ROWS * width + col;
  float s = 0.f, s2 = 0.f;
  for (int r = threadIdx.x; r < N_ROWS; r += blockDim.x) {
    float v = base[(size_t)r * width];
    s += v; s2 += v * v;
  }
  __shared__ float sh[256], sh2[256];
  sh[threadIdx.x] = s; sh2[threadIdx.x] = s2;
  __syncthreads();
  for (int o = blockDim.x >> 1; o > 0; o >>= 1) {
    if ((int)threadIdx.x < o) { sh[threadIdx.x] += sh[threadIdx.x + o];
                                sh2[threadIdx.x] += sh2[threadIdx.x + o]; }
    __syncthreads();
  }
  if (threadIdx.x == 0) {
    float mean = sh[0] / (float)N_ROWS;
    float var  = sh2[0] / (float)N_ROWS - mean * mean;
    float sc   = g[col] * rsqrtf(var + BN_EPS);
    scale[blockIdx.x] = sc;
    shift[blockIdx.x] = bt[col] - mean * sc;
  }
}

// ---------------- Layer 1: H1 = tanh(BN0(H0) @ W1^T + b1) --------------------------
__global__ void k_gemm1(const float* __restrict__ H0,
                        const float* __restrict__ scale0, const float* __restrict__ shift0,
                        const float* __restrict__ W1, const float* __restrict__ b1,
                        float* __restrict__ H1v) {
  __shared__ float row[HID1];
  const int r = blockIdx.x;
  const int batch = (r >= N_ROWS);
  const float* h  = H0 + (size_t)r * HID1;
  const float* sc = scale0 + batch * HID1;
  const float* sf = shift0 + batch * HID1;
  for (int i = threadIdx.x; i < HID1; i += blockDim.x)
    row[i] = fmaf(h[i], sc[i], sf[i]);
  __syncthreads();
  const int e = threadIdx.x;                 // 64 threads = 64 output features
  const float* w = W1 + (size_t)e * HID1;
  float acc = b1[e];
#pragma unroll 8
  for (int c = 0; c < HID1; ++c) acc = fmaf(row[c], w[c], acc);
  H1v[(size_t)r * HID2 + e] = tanhf(acc);
}

// ------- BN1 apply, f16 convert, row squared-norms (from rounded f16 values) -------
__global__ void k_final(const float* __restrict__ H1v,
                        const float* __restrict__ scale1, const float* __restrict__ shift1,
                        _Float16* __restrict__ feat, float* __restrict__ nrm) {
  const int r = blockIdx.x;
  const int batch = (r >= N_ROWS);
  const int e = threadIdx.x;
  float v = fmaf(H1v[(size_t)r * HID2 + e], scale1[batch * HID2 + e], shift1[batch * HID2 + e]);
  _Float16 hv = (_Float16)v;
  feat[(size_t)r * HID2 + e] = hv;
  float f = (float)hv;
  __shared__ float sh[HID2];
  sh[e] = f * f;
  __syncthreads();
  for (int o = HID2 >> 1; o > 0; o >>= 1) {
    if (e < o) sh[e] += sh[e + o];
    __syncthreads();
  }
  if (e == 0) nrm[r] = sh[0];
}

// ---- Fused cdist + exp(-d) + (w @ one_hot) via WMMA; one wave per 16-row tile -----
__global__ void __launch_bounds__(32)
k_dist(const _Float16* __restrict__ feat, const float* __restrict__ nrm,
       const int* __restrict__ yn, float* __restrict__ partial) {
  __shared__ alignas(16) _Float16 lw[16 * 32];  // w tile, rows x 32 cols
  __shared__ int ly[32];                        // class labels for 32 cols

  const int lane = threadIdx.x;
  const int rowTile = blockIdx.x;               // 0..511
  const int chunk   = blockIdx.y;               // 0..SPLIT-1
  const int m0   = rowTile * 16;
  const int half = (lane >= 16) ? 1 : 0;
  const int l15  = lane & 15;

  // A tiles: q rows m0..m0+15, K=64 split into two 16x32 f16 WMMA A operands.
  // A operand, contiguous k order used here (same as B): lane L holds M=L&15; half-wave K-base 0/16; chunks
  // [kb,kb+8) and [kb+8,kb+16) per operand.
  const _Float16* qrow = feat + (size_t)(m0 + l15) * HID2;
  const int kb = half ? 16 : 0;
  v8h a0 = *(const v8h*)(qrow + kb);
  v8h a1 = *(const v8h*)(qrow + kb + 8);
  v8h a2 = *(const v8h*)(qrow + 32 + kb);
  v8h a3 = *(const v8h*)(qrow + 32 + kb + 8);
  v16h A_lo, A_hi;
#pragma unroll
  for (int i = 0; i < 8; ++i) {
    A_lo[i] = a0[i]; A_lo[8 + i] = a1[i];
    A_hi[i] = a2[i]; A_hi[8 + i] = a3[i];
  }

  // ||q||^2 for the 8 C-matrix rows this lane owns (M = half*8 + v).
  float qv[8];
  {
    const float* qq = nrm + m0 + half * 8;
#pragma unroll
    for (int v = 0; v < 8; ++v) qv[v] = qq[v];
  }

  v8f outC = {0.f, 0.f, 0.f, 0.f, 0.f, 0.f, 0.f, 0.f};  // class accumulators (16x16 C)

  const _Float16* kfeat = feat + (size_t)N_ROWS * HID2;
  const float*    kk    = nrm + N_ROWS;
  const int colsPerChunk = M_ROWS / SPLIT;      // 512
  const int cbase = chunk * colsPerChunk;
  const int kb2 = half ? 16 : 0;                // B layout: contiguous 16 K per half-wave

  for (int g = 0; g < colsPerChunk / 32; ++g) {
    const int colbase = cbase + g * 32;
    ly[lane] = yn[colbase + lane];
    __syncthreads();

#pragma unroll
    for (int t = 0; t < 2; ++t) {
      const int ncol = colbase + t * 16 + l15;
      const _Float16* krow = kfeat + (size_t)ncol * HID2;
      v8h b0v = *(const v8h*)(krow + kb2);
      v8h b1v = *(const v8h*)(krow + kb2 + 8);
      v8h b2v = *(const v8h*)(krow + 32 + kb2);
      v8h b3v = *(const v8h*)(krow + 32 + kb2 + 8);
      v16h B_lo, B_hi;
#pragma unroll
      for (int i = 0; i < 8; ++i) {
        B_lo[i] = b0v[i]; B_lo[8 + i] = b1v[i];
        B_hi[i] = b2v[i]; B_hi[8 + i] = b3v[i];
      }
      const float kkv = kk[ncol];

      // S = q_tile @ k_tile^T over K=64 (two k-steps), f32 accumulate.
      v8f S = {0.f, 0.f, 0.f, 0.f, 0.f, 0.f, 0.f, 0.f};
      S = __builtin_amdgcn_wmma_f32_16x16x32_f16(false, A_lo, false, B_lo,
                                                 (short)0, S, false, false);
      S = __builtin_amdgcn_wmma_f32_16x16x32_f16(false, A_hi, false, B_hi,
                                                 (short)0, S, false, false);

      // w = exp(-sqrt(max(||q||^2 + ||k||^2 - 2S, 1e-12))); stage f16 tile in LDS.
      // __builtin_amdgcn_sqrtf -> single v_sqrt_f32 (TRANS, co-executes with WMMA).
#pragma unroll
      for (int v = 0; v < 8; ++v) {
        float d2   = qv[v] + kkv - 2.0f * S[v];
        float dist = __builtin_amdgcn_sqrtf(fmaxf(d2, 1e-12f));
        float w    = __expf(-dist);
        int m = half * 8 + v;
        lw[m * 32 + t * 16 + l15] = (_Float16)w;
      }
    }
    __syncthreads();

    // One-hot B tile (padded 10 -> 16 classes): B2[k][c] = (y[colbase+k] == c).
    v16h B2;
#pragma unroll
    for (int i = 0; i < 16; ++i)
      B2[i] = (ly[kb2 + i] == l15) ? (_Float16)1.0f : (_Float16)0.0f;

    // Repack w tile from C layout to A layout via LDS (two ds_load_b128).
    v16h A2;
    {
      v8h c0 = *(const v8h*)(lw + l15 * 32 + kb);
      v8h c1 = *(const v8h*)(lw + l15 * 32 + kb + 8);
#pragma unroll
      for (int i = 0; i < 8; ++i) { A2[i] = c0[i]; A2[8 + i] = c1[i]; }
    }

    // outC += w_tile @ one_hot  (class sums; total = sum over the 10 classes)
    outC = __builtin_amdgcn_wmma_f32_16x16x32_f16(false, A2, false, B2,
                                                  (short)0, outC, false, false);
    __syncthreads();
  }

  float* dst = partial + ((size_t)(rowTile * SPLIT) + chunk) * 256;
#pragma unroll
  for (int v = 0; v < 8; ++v) {
    int m = half * 8 + v;
    dst[m * 16 + l15] = outC[v];
  }
}

// ------------- Reduce column-chunk partials, normalize, clip, emit [N,10] ----------
__global__ void k_combine(const float* __restrict__ partial, float* __restrict__ out) {
  const int r = blockIdx.x * blockDim.x + threadIdx.x;
  if (r >= N_ROWS) return;
  const int rowTile = r >> 4, mrow = r & 15;
  float cls[NCLS];
#pragma unroll
  for (int c = 0; c < NCLS; ++c) cls[c] = 0.f;
  for (int s = 0; s < SPLIT; ++s) {
    const float* p = partial + ((size_t)(rowTile * SPLIT) + s) * 256 + mrow * 16;
#pragma unroll
    for (int c = 0; c < NCLS; ++c) cls[c] += p[c];
  }
  float tot = 0.f;
#pragma unroll
  for (int c = 0; c < NCLS; ++c) tot += cls[c];
  float inv = 1.0f / tot;
  float* o = out + (size_t)r * NCLS;
#pragma unroll
  for (int c = 0; c < NCLS; ++c) o[c] = fminf(fmaxf(cls[c] * inv, 0.f), 1.f);
}

extern "C" void kernel_launch(void* const* d_in, const int* in_sizes, int n_in,
                              void* d_out, int out_size, void* d_ws, size_t ws_size,
                              hipStream_t stream) {
  const float* x   = (const float*)d_in[0];
  const float* xn  = (const float*)d_in[1];
  const int*   yn  = (const int*)d_in[2];
  const float* W0  = (const float*)d_in[3];
  const float* b0  = (const float*)d_in[4];
  const float* g0  = (const float*)d_in[5];
  const float* bt0 = (const float*)d_in[6];
  const float* W1  = (const float*)d_in[7];
  const float* b1  = (const float*)d_in[8];
  const float* g1  = (const float*)d_in[9];
  const float* bt1 = (const float*)d_in[10];

  char* ws = (char*)d_ws;
  float*    H0      = (float*)ws;      ws += (size_t)TOT * HID1 * sizeof(float);
  float*    H1v     = (float*)ws;      ws += (size_t)TOT * HID2 * sizeof(float);
  float*    scale0  = (float*)ws;      ws += 2 * HID1 * sizeof(float);
  float*    shift0  = (float*)ws;      ws += 2 * HID1 * sizeof(float);
  float*    scale1  = (float*)ws;      ws += 2 * HID2 * sizeof(float);
  float*    shift1  = (float*)ws;      ws += 2 * HID2 * sizeof(float);
  _Float16* feat    = (_Float16*)ws;   ws += (size_t)TOT * HID2 * sizeof(_Float16);
  float*    nrm     = (float*)ws;      ws += (size_t)TOT * sizeof(float);
  float*    partial = (float*)ws;      ws += (size_t)(N_ROWS / 16) * SPLIT * 256 * sizeof(float);

  k_gemm0<<<TOT, HID1, 0, stream>>>(x, xn, W0, b0, H0);
  k_stats<<<2 * HID1, 256, 0, stream>>>(H0, HID1, g0, bt0, scale0, shift0);
  k_gemm1<<<TOT, HID2, 0, stream>>>(H0, scale0, shift0, W1, b1, H1v);
  k_stats<<<2 * HID2, 256, 0, stream>>>(H1v, HID2, g1, bt1, scale1, shift1);
  k_final<<<TOT, HID2, 0, stream>>>(H1v, scale1, shift1, feat, nrm);
  k_dist<<<dim3(N_ROWS / 16, SPLIT), 32, 0, stream>>>(feat, nrm, yn, partial);
  k_combine<<<(N_ROWS + 255) / 256, 256, 0, stream>>>(partial, (float*)d_out);
}
